// PAM_Module_15693810499975
// MI455X (gfx1250) — matrix operand order repaired and checked by reading, not compiled, not run on hardware
//
#include <hip/hip_runtime.h>
#include <math.h>

// ---------------------------------------------------------------------------
// PAM (position attention) + SELayer fused pipeline for MI455X (gfx1250).
// B=8, C=128, H=W=48 -> N=2304.  Big GEMMs: v_wmma_f32_16x16x32_f16.
// K/V tiles staged into LDS via Tensor Data Mover (tensor_load_to_lds),
// double-buffered and synchronized with s_wait_tensorcnt.
// ---------------------------------------------------------------------------

#define BB   8
#define CCH  128
#define HH   48
#define WW   48
#define NPIX (HH * WW)          // 2304
#define HID  8                  // C / RED

typedef __attribute__((ext_vector_type(16))) _Float16 v16h;
typedef __attribute__((ext_vector_type(8)))  _Float16 v8h;
typedef __attribute__((ext_vector_type(8)))  float    v8f;
typedef __attribute__((ext_vector_type(4)))  unsigned int u32x4;
typedef __attribute__((ext_vector_type(8)))  int          i32x8;
typedef __attribute__((ext_vector_type(4)))  int          i32x4;

#if __has_builtin(__builtin_amdgcn_tensor_load_to_lds) && \
    __has_builtin(__builtin_amdgcn_s_wait_tensorcnt)
#define HAVE_TDM 1
#else
#define HAVE_TDM 0
#endif

// ---------------------------------------------------------------------------
// WMMA helper (16x16x32 f16 -> f32).  Layouts per CDNA5 ISA 7.12.2.
// ---------------------------------------------------------------------------
__device__ __forceinline__ v8f wmma_f16(v16h a, v16h b, v8f c) {
  return __builtin_amdgcn_wmma_f32_16x16x32_f16(
      /*neg_a=*/false, a, /*neg_b=*/false, b,
      /*c_mod=*/(short)0, c, /*reuse_a=*/false, /*reuse_b=*/false);
}

// A fragment from row-major M[row][k], leading dim ld.
// Lane holds row M = lane&15; halves K {0..7,16..23} (lanes 0-15) or
// K {8..15,24..31} (lanes 16-31).
__device__ __forceinline__ v16h load_afrag(const _Float16* base,
                                           int ld, int row0, int k0, int lane) {
  int row  = row0 + (lane & 15);
  int koff = (lane & 16) ? 8 : 0;
  const _Float16* p = base + (size_t)row * ld + k0;
  v8h lo = *(const v8h*)(p + koff);
  v8h hi = *(const v8h*)(p + 16 + koff);
  v16h r;
#pragma unroll
  for (int t = 0; t < 8; ++t) { r[t] = lo[t]; r[8 + t] = hi[t]; }
  return r;
}

// B fragment where columns are contiguous in K: T[col][k], ld = K-stride.
// Lane holds column N = col0 + (lane&15); halves K {0..7,16..23} (lanes 0-15)
// or K {8..15,24..31} (lanes 16-31).
__device__ __forceinline__ v16h load_bfrag(const _Float16* base,
                                           int ld, int col0, int k0, int lane) {
  int col  = col0 + (lane & 15);
  int koff = (lane & 16) ? 8 : 0;
  const _Float16* p = base + (size_t)col * ld + k0;
  v8h lo = *(const v8h*)(p + koff);
  v8h hi = *(const v8h*)(p + 16 + koff);
  v16h r;
#pragma unroll
  for (int t = 0; t < 8; ++t) { r[t] = lo[t]; r[8 + t] = hi[t]; }
  return r;
}

#if HAVE_TDM
// ---------------------------------------------------------------------------
// Tensor Data Mover: issue a 2-D tile load (global -> LDS), data_size = 2B.
// Descriptor layout per CDNA5 ISA §8.3-8.5; groups 2/3 zeroed (<=2D tensor).
// clang-23 toolchain: 6-arg builtin (extra i32x8 group before cpol).
// ---------------------------------------------------------------------------
__device__ __forceinline__ void tdm_load_tile(
    unsigned lds_addr, const void* gptr,
    unsigned tensor_d0, unsigned tensor_d1,
    unsigned tile_d0, unsigned tile_d1,
    unsigned long long stride0) {
  unsigned long long ga = (unsigned long long)gptr;
  u32x4 g0;
  g0[0] = 1u;                                              // count=1, user D#
  g0[1] = lds_addr;                                        // lds_addr[31:0]
  g0[2] = (unsigned)(ga & 0xFFFFFFFFull);                  // global_addr lo
  g0[3] = (unsigned)((ga >> 32) & 0x1FFFFFFull) | (2u << 30);  // hi + type=2
  i32x8 g1;
  g1[0] = (int)(1u << 16);                                 // data_size = 2B
  g1[1] = (int)((tensor_d0 & 0xFFFFu) << 16);              // dim0[15:0]
  g1[2] = (int)(((tensor_d0 >> 16) & 0xFFFFu) |
                ((tensor_d1 & 0xFFFFu) << 16));            // dim0[31:16],dim1
  g1[3] = (int)(((tensor_d1 >> 16) & 0xFFFFu) |
                ((tile_d0 & 0xFFFFu) << 16));              // dim1[31:16],tile0
  g1[4] = (int)(tile_d1 & 0xFFFFu);                        // tile1, tile2=0
  g1[5] = (int)(stride0 & 0xFFFFFFFFull);                  // dim0_stride lo
  g1[6] = (int)((stride0 >> 32) & 0xFFFFull);              // dim0_stride hi
  g1[7] = 0;
  i32x4 z4 = {0, 0, 0, 0};
  i32x8 z8 = {0, 0, 0, 0, 0, 0, 0, 0};
  __builtin_amdgcn_tensor_load_to_lds(g0, g1, z4, z4, z8, 0);
}
#endif

// ---------------------------------------------------------------------------
// Kernel 1: Wq/Wk/Wv (f32) -> packed f16 [3][C][C].
// ---------------------------------------------------------------------------
__global__ void prep_w_kernel(const float* __restrict__ Wq,
                              const float* __restrict__ Wk,
                              const float* __restrict__ Wv,
                              _Float16* __restrict__ Wh) {
  int i = blockIdx.x * blockDim.x + threadIdx.x;
  if (i >= 3 * CCH * CCH) return;
  const float* src = (i < CCH * CCH) ? Wq : (i < 2 * CCH * CCH ? Wk : Wv);
  Wh[i] = (_Float16)src[i % (CCH * CCH)];
}

// ---------------------------------------------------------------------------
// Kernel 2: x (B,C,N f32) -> xT (B,N,C f16).
// ---------------------------------------------------------------------------
__global__ void prep_x_kernel(const float* __restrict__ x,
                              _Float16* __restrict__ xT) {
  int idx = blockIdx.x * blockDim.x + threadIdx.x;
  if (idx >= BB * CCH * NPIX) return;
  int n = idx % NPIX;
  int t = idx / NPIX;
  int c = t % CCH;
  int b = t / CCH;
  xT[((size_t)b * NPIX + n) * CCH + c] = (_Float16)x[idx];
}

// ---------------------------------------------------------------------------
// Kernel 3: SE gates.  y[b,c] = sigmoid(w2 @ relu(w1 @ mean_n(x)))
// ---------------------------------------------------------------------------
__global__ __launch_bounds__(CCH) void se_kernel(const float* __restrict__ x,
                                                 const float* __restrict__ w1,
                                                 const float* __restrict__ w2,
                                                 float* __restrict__ y) {
  __shared__ float smean[CCH];
  __shared__ float shid[HID];
  int b = blockIdx.x;
  int c = threadIdx.x;
  const float* xb = x + ((size_t)b * CCH + c) * NPIX;
  float s = 0.f;
  for (int n = 0; n < NPIX; ++n) s += xb[n];
  smean[c] = s * (1.0f / NPIX);
  __syncthreads();
  if (c < HID) {
    float a = 0.f;
    for (int j = 0; j < CCH; ++j) a += w1[c * CCH + j] * smean[j];
    shid[c] = fmaxf(a, 0.f);
  }
  __syncthreads();
  float a = 0.f;
#pragma unroll
  for (int t = 0; t < HID; ++t) a += w2[c * HID + t] * shid[t];
  y[b * CCH + c] = 1.0f / (1.0f + __expf(-a));
}

// ---------------------------------------------------------------------------
// Kernel 4: QKV projections via WMMA (8 waves, one 16-channel tile each).
// qT,kT -> (B,N,C) f16;  v -> (B,C,N) f16.
// ---------------------------------------------------------------------------
__global__ __launch_bounds__(256) void qkv_kernel(
    const _Float16* __restrict__ Wh,
    const float* __restrict__ bq, const float* __restrict__ bk,
    const float* __restrict__ bv,
    const _Float16* __restrict__ xT,
    _Float16* __restrict__ qT, _Float16* __restrict__ kT,
    _Float16* __restrict__ vM) {
  int lane = threadIdx.x & 31;
  int wave = threadIdx.x >> 5;
  int ntpb = NPIX / 16;
  int b  = blockIdx.x / ntpb;
  int n0 = (blockIdx.x % ntpb) * 16;
  int o0 = wave * 16;

  const _Float16* xTb = xT + (size_t)b * NPIX * CCH;
  v16h bX[4];
#pragma unroll
  for (int kc = 0; kc < 4; ++kc) bX[kc] = load_bfrag(xTb, CCH, n0, kc * 32, lane);

  int rofs = (lane & 16) ? 8 : 0;
  int n    = n0 + (lane & 15);
  const float* biases[3] = {bq, bk, bv};

#pragma unroll
  for (int p = 0; p < 3; ++p) {
    const _Float16* W = Wh + p * CCH * CCH;
    v8f acc = {};
#pragma unroll
    for (int kc = 0; kc < 4; ++kc)
      acc = wmma_f16(load_afrag(W, CCH, o0, kc * 32, lane), bX[kc], acc);

    const float* bias = biases[p];
    if (p < 2) {
      v8h h;
#pragma unroll
      for (int r = 0; r < 8; ++r)
        h[r] = (_Float16)(acc[r] + bias[o0 + rofs + r]);
      _Float16* dst = (p == 0 ? qT : kT) +
                      (((size_t)b * NPIX + n) * CCH + o0 + rofs);
      *(v8h*)dst = h;
    } else {
#pragma unroll
      for (int r = 0; r < 8; ++r) {
        int o = o0 + rofs + r;
        vM[((size_t)b * CCH + o) * NPIX + n] = (_Float16)(acc[r] + bias[o]);
      }
    }
  }
}

// ---------------------------------------------------------------------------
// Kernel 5: flash-attention + fused epilogue.
// Block = 4 waves; wave w owns rows [i0, i0+16).  j streamed in tiles of 32.
// K tile (32 x 128 halves, contiguous 8KB) and V tile (128 rows x 32 halves,
// stride N) staged in LDS by the TDM, double-buffered.
// ---------------------------------------------------------------------------
__global__ __launch_bounds__(128) void attn_kernel(
    const _Float16* __restrict__ qT,   // [B][N][C]
    const _Float16* __restrict__ kT,   // [B][N][C]
    const _Float16* __restrict__ vM,   // [B][C][N]
    const float* __restrict__ x,       // [B][C][N]
    const float* __restrict__ yse,     // [B][C]
    const float* __restrict__ gamma,   // [1]
    float* __restrict__ out) {         // [B][C][N]
  __shared__ _Float16 ktile[2][32 * CCH];   // [j][c]
  __shared__ _Float16 vtile[2][CCH * 32];   // [c][j]
  __shared__ _Float16 pbuf[4][16 * 32];     // per-wave P staging

  int lane = threadIdx.x & 31;
  int wave = threadIdx.x >> 5;
  int ibpb = NPIX / 64;
  int b  = blockIdx.x / ibpb;
  int i0 = (blockIdx.x % ibpb) * 64 + wave * 16;

  const _Float16* qTb = qT + (size_t)b * NPIX * CCH;
  const _Float16* kTb = kT + (size_t)b * NPIX * CCH;
  const _Float16* vb  = vM + (size_t)b * CCH * NPIX;

  // Q A-fragments invariant over j loop.
  v16h aQ[4];
#pragma unroll
  for (int kc = 0; kc < 4; ++kc) aQ[kc] = load_afrag(qTb, CCH, i0, kc * 32, lane);

  v8f acc[8];
#pragma unroll
  for (int ct = 0; ct < 8; ++ct) acc[ct] = v8f{};
  float mrun[8], lrun[8];
#pragma unroll
  for (int r = 0; r < 8; ++r) { mrun[r] = -3.0e38f; lrun[r] = 0.f; }

  int rofs = (lane & 16) ? 8 : 0;
  int col  = lane & 15;
  _Float16* pb = pbuf[wave];

#if HAVE_TDM
  unsigned klds[2] = {(unsigned)(size_t)&ktile[0][0], (unsigned)(size_t)&ktile[1][0]};
  unsigned vlds[2] = {(unsigned)(size_t)&vtile[0][0], (unsigned)(size_t)&vtile[1][0]};
  // Prologue: stage tile j0 = 0 into buffer 0.
  if (wave == 0)
    tdm_load_tile(klds[0], kTb, 32 * CCH, 1, 32 * CCH, 1, 32 * CCH);
  else if (wave == 1)
    tdm_load_tile(vlds[0], vb, NPIX, CCH, 32, CCH, NPIX);
#endif
  int buf = 0;

  for (int j0 = 0; j0 < NPIX; j0 += 32) {
#if HAVE_TDM
    // Issue next tile into buf^1, then wait for current tile (in-order TDM).
    if (wave == 0) {
      if (j0 + 32 < NPIX) {
        tdm_load_tile(klds[buf ^ 1], kTb + (size_t)(j0 + 32) * CCH,
                      32 * CCH, 1, 32 * CCH, 1, 32 * CCH);
        __builtin_amdgcn_s_wait_tensorcnt(1);
      } else {
        __builtin_amdgcn_s_wait_tensorcnt(0);
      }
    } else if (wave == 1) {
      if (j0 + 32 < NPIX) {
        tdm_load_tile(vlds[buf ^ 1], vb + (j0 + 32), NPIX, CCH, 32, CCH, NPIX);
        __builtin_amdgcn_s_wait_tensorcnt(1);
      } else {
        __builtin_amdgcn_s_wait_tensorcnt(0);
      }
    }
    __syncthreads();                 // current tile visible to all waves
#else
    // Fallback: cooperative staging of tile j0 into buffer 0.
    __syncthreads();                 // previous iteration finished reading
    {
      int t = threadIdx.x;
#pragma unroll
      for (int it = 0; it < 4; ++it) {       // K tile: 512 chunks of 16B
        int ch = it * 128 + t;
        int row = ch >> 4, off = (ch & 15) * 8;
        *(v8h*)&ktile[0][row * CCH + off] =
            *(const v8h*)(kTb + (size_t)(j0 + row) * CCH + off);
      }
#pragma unroll
      for (int it = 0; it < 4; ++it) {       // V tile: 512 chunks of 16B
        int ch = it * 128 + t;
        int row = ch >> 2, off = (ch & 3) * 8;
        *(v8h*)&vtile[0][row * 32 + off] =
            *(const v8h*)(vb + (size_t)row * NPIX + j0 + off);
      }
    }
    __syncthreads();
#endif

    const _Float16* kt = &ktile[buf][0];
    const _Float16* vt = &vtile[buf][0];

    // ---- S = Q^T K for 16 rows x 32 cols (B frags from LDS) ------------
    v8f s0 = {}, s1 = {};
#pragma unroll
    for (int kc = 0; kc < 4; ++kc) {
      v16h bK0 = load_bfrag(kt, CCH, 0,  kc * 32, lane);
      v16h bK1 = load_bfrag(kt, CCH, 16, kc * 32, lane);
      s0 = wmma_f16(aQ[kc], bK0, s0);
      s1 = wmma_f16(aQ[kc], bK1, s1);
    }

    // ---- online softmax ------------------------------------------------
    float scalef[8];
#pragma unroll
    for (int r = 0; r < 8; ++r) {
      float tm = fmaxf(s0[r], s1[r]);
      tm = fmaxf(tm, __shfl_xor(tm, 1, 32));
      tm = fmaxf(tm, __shfl_xor(tm, 2, 32));
      tm = fmaxf(tm, __shfl_xor(tm, 4, 32));
      tm = fmaxf(tm, __shfl_xor(tm, 8, 32));
      float mnew = fmaxf(mrun[r], tm);
      float sc   = __expf(mrun[r] - mnew);
      mrun[r]    = mnew;
      float p0 = __expf(s0[r] - mnew);
      float p1 = __expf(s1[r] - mnew);
      s0[r] = p0;
      s1[r] = p1;
      float rs = p0 + p1;
      rs += __shfl_xor(rs, 1, 32);
      rs += __shfl_xor(rs, 2, 32);
      rs += __shfl_xor(rs, 4, 32);
      rs += __shfl_xor(rs, 8, 32);
      lrun[r] = lrun[r] * sc + rs;
      scalef[r] = sc;
    }
#pragma unroll
    for (int ct = 0; ct < 8; ++ct)
#pragma unroll
      for (int r = 0; r < 8; ++r) acc[ct][r] *= scalef[r];

    // ---- repack P (C-layout f32) -> A-fragment layout via LDS ----------
#pragma unroll
    for (int r = 0; r < 8; ++r) {
      pb[(r + rofs) * 32 + col]      = (_Float16)s0[r];
      pb[(r + rofs) * 32 + col + 16] = (_Float16)s1[r];
    }
    __syncthreads();
    v16h aP = load_afrag(pb, 32, 0, 0, lane);
    __syncthreads();

    // ---- O += P * V^T over 8 channel tiles (B frags from LDS) ----------
#pragma unroll
    for (int ct = 0; ct < 8; ++ct) {
      v16h bV = load_bfrag(vt, 32, ct * 16, 0, lane);
      acc[ct] = wmma_f16(aP, bV, acc[ct]);
    }

#if HAVE_TDM
    __syncthreads();                 // all reads of tile[buf] done before refill
    buf ^= 1;
#endif
  }

  // ---- epilogue: normalize, fuse gamma*out + x*se ----------------------
  float g = gamma[0];
  float inv[8];
#pragma unroll
  for (int r = 0; r < 8; ++r) inv[r] = 1.0f / lrun[r];

#pragma unroll
  for (int ct = 0; ct < 8; ++ct) {
    int c = ct * 16 + col;
    float gate = yse[b * CCH + c];
    const float* xrow = x + ((size_t)b * CCH + c) * NPIX;
    float* orow = out + ((size_t)b * CCH + c) * NPIX;
#pragma unroll
    for (int r = 0; r < 8; ++r) {
      int i = i0 + rofs + r;
      orow[i] = g * acc[ct][r] * inv[r] + xrow[i] * gate;
    }
  }
}

// ---------------------------------------------------------------------------
// Host-side launcher.
// Inputs: 0:x 1:Wq 2:bq 3:Wk 4:bk 5:Wv 6:bv 7:se_w1 8:se_w2 9:gamma
// ---------------------------------------------------------------------------
extern "C" void kernel_launch(void* const* d_in, const int* in_sizes, int n_in,
                              void* d_out, int out_size, void* d_ws, size_t ws_size,
                              hipStream_t stream) {
  const float* x     = (const float*)d_in[0];
  const float* Wq    = (const float*)d_in[1];
  const float* bq    = (const float*)d_in[2];
  const float* Wk    = (const float*)d_in[3];
  const float* bk    = (const float*)d_in[4];
  const float* Wv    = (const float*)d_in[5];
  const float* bv    = (const float*)d_in[6];
  const float* se_w1 = (const float*)d_in[7];
  const float* se_w2 = (const float*)d_in[8];
  const float* gamma = (const float*)d_in[9];
  float* out = (float*)d_out;

  char* ws = (char*)d_ws;
  const size_t BNC = (size_t)BB * NPIX * CCH;
  size_t off = 0;
  _Float16* Wh = (_Float16*)(ws + off); off += 3 * CCH * CCH * sizeof(_Float16);
  off = (off + 255) & ~(size_t)255;
  _Float16* xT = (_Float16*)(ws + off); off += BNC * sizeof(_Float16);
  _Float16* qT = (_Float16*)(ws + off); off += BNC * sizeof(_Float16);
  _Float16* kT = (_Float16*)(ws + off); off += BNC * sizeof(_Float16);
  _Float16* vM = (_Float16*)(ws + off); off += BNC * sizeof(_Float16);
  float*    yg = (float*)(ws + off);    off += BB * CCH * sizeof(float);

  prep_w_kernel<<<(3 * CCH * CCH + 255) / 256, 256, 0, stream>>>(Wq, Wk, Wv, Wh);
  prep_x_kernel<<<(int)((BNC + 255) / 256), 256, 0, stream>>>(x, xT);
  se_kernel<<<BB, CCH, 0, stream>>>(x, se_w1, se_w2, yg);
  qkv_kernel<<<BB * (NPIX / 16), 256, 0, stream>>>(Wh, bq, bk, bv, xT, qT, kT, vM);
  attn_kernel<<<BB * (NPIX / 64), 128, 0, stream>>>(qT, kT, vM, x, yg, gamma, out);
}
